// GenModel_36215164240676
// MI455X (gfx1250) — compile-verified, matrix operand order checked by reading, not run on hardware
//
#include <hip/hip_runtime.h>

typedef __attribute__((ext_vector_type(16))) __bf16 v16bf;
typedef __attribute__((ext_vector_type(8)))  __bf16 v8bf;
typedef __attribute__((ext_vector_type(8)))  float  v8f;

__device__ __forceinline__ unsigned short f2bf(float f) {
    union { float f; unsigned u; } v; v.f = f;
    unsigned r = v.u + 0x7FFFu + ((v.u >> 16) & 1u);   // round-to-nearest-even
    return (unsigned short)(r >> 16);
}

// ---------------------------------------------------------------------------
// Generic tiled GEMM:  C = relu(A[M x K](bf16) * B[N x K](f32,row-major)^T + bias)
// Output row m -> C[( (m/P)*rowStride + rowBase + (m%P) ) * ldc + n]
// Tile: 128(M) x 128(N) x 32(K).  256 threads = 8 waves (2 x 4), each wave
// computes a 64x32 sub-tile via 4x2 v_wmma_f32_16x16x32_bf16 accumulators.
// ---------------------------------------------------------------------------
#define TM 128
#define TN 128
#define TK 32

__global__ __launch_bounds__(256)
void gemm_bf16_relu(const unsigned short* __restrict__ A,
                    const float*          __restrict__ B,
                    const float*          __restrict__ bias,
                    float*                __restrict__ C,
                    int M, int N, int K,
                    int P, int rowStride, int rowBase, int ldc)
{
    __shared__ __align__(16) unsigned short As[TM * TK];
    __shared__ __align__(16) unsigned short Bs[TN * TK];

    const int t     = threadIdx.x;
    const int wave  = t >> 5;
    const int lane  = t & 31;
    const int waveM = wave >> 2;       // 0..1
    const int waveN = wave & 3;        // 0..3
    const int hf    = lane >> 4;       // 0/1 : K-half select per ISA layout
    const int l15   = lane & 15;

    const int blockRow = blockIdx.x * TM;
    const int blockCol = blockIdx.y * TN;

    const v8f vzero = {0.f,0.f,0.f,0.f,0.f,0.f,0.f,0.f};
    v8f acc[4][2];
    #pragma unroll
    for (int i = 0; i < 4; ++i)
        #pragma unroll
        for (int j = 0; j < 2; ++j) acc[i][j] = vzero;

    for (int kt = 0; kt < K; kt += TK) {
        // --- A tile: 128x32 bf16 from workspace (already bf16), 2 x b128/thread
        #pragma unroll
        for (int i = 0; i < 2; ++i) {
            int off = (t + i * 256) * 8;           // bf16 element offset in tile
            int r   = off >> 5, cc = off & 31;
            int gr  = blockRow + r;
            uint4 v; v.x = v.y = v.z = v.w = 0u;
            if (gr < M) v = *(const uint4*)(A + (size_t)gr * K + kt + cc);
            *(uint4*)(&As[r * TK + cc]) = v;
        }
        // --- B tile: 128x32 f32 from HBM -> bf16 into LDS, 4 x b128/thread
        #pragma unroll
        for (int i = 0; i < 4; ++i) {
            int off = (t + i * 256) * 4;           // f32 element offset in tile
            int r   = off >> 5, cc = off & 31;
            const float4 v = *(const float4*)(B + (size_t)(blockCol + r) * K + kt + cc);
            ushort4 bv;
            bv.x = f2bf(v.x); bv.y = f2bf(v.y); bv.z = f2bf(v.z); bv.w = f2bf(v.w);
            *(ushort4*)(&Bs[r * TK + cc]) = bv;
        }
        __syncthreads();

        // --- fragment loads (two contiguous 16B LDS chunks per fragment) ---
        v16bf afr[4], bfr[2];
        #pragma unroll
        for (int mi = 0; mi < 4; ++mi) {
            int rowIdx = waveM * 64 + mi * 16 + l15;
            v8bf lo = *(const v8bf*)(&As[rowIdx * TK + hf * 8]);
            v8bf hi = *(const v8bf*)(&As[rowIdx * TK + 16 + hf * 8]);
            afr[mi] = __builtin_shufflevector(lo, hi, 0,1,2,3,4,5,6,7,8,9,10,11,12,13,14,15);
        }
        #pragma unroll
        for (int ni = 0; ni < 2; ++ni) {
            int colIdx = waveN * 32 + ni * 16 + l15;
            v8bf lo = *(const v8bf*)(&Bs[colIdx * TK + hf * 8]);
            v8bf hi = *(const v8bf*)(&Bs[colIdx * TK + 16 + hf * 8]);
            bfr[ni] = __builtin_shufflevector(lo, hi, 0,1,2,3,4,5,6,7,8,9,10,11,12,13,14,15);
        }

        #pragma unroll
        for (int mi = 0; mi < 4; ++mi)
            #pragma unroll
            for (int ni = 0; ni < 2; ++ni)
                acc[mi][ni] = __builtin_amdgcn_wmma_f32_16x16x32_bf16(
                    false, afr[mi], false, bfr[ni],
                    (short)0, acc[mi][ni], false, false);
        __syncthreads();
    }

    // --- epilogue: bias + relu, scatter rows into concatenated output ---
    #pragma unroll
    for (int ni = 0; ni < 2; ++ni) {
        int gn   = blockCol + waveN * 32 + ni * 16 + l15;
        float bv = bias[gn];
        #pragma unroll
        for (int mi = 0; mi < 4; ++mi) {
            #pragma unroll
            for (int r = 0; r < 8; ++r) {
                int gm = blockRow + waveM * 64 + mi * 16 + r + 8 * hf;
                if (gm < M) {
                    float val = acc[mi][ni][r] + bv;
                    val = val > 0.f ? val : 0.f;
                    int b = gm / P, p = gm - b * P;
                    C[(size_t)(b * rowStride + rowBase + p) * ldc + gn] = val;
                }
            }
        }
    }
}

// ---------------------------------------------------------------------------
// im2col / convert kernels (f32 -> bf16 patch matrices)
// ---------------------------------------------------------------------------
__global__ void pk1_kernel(const float* __restrict__ x, unsigned short* __restrict__ p1, int total) {
    int idx = blockIdx.x * blockDim.x + threadIdx.x;
    if (idx >= total) return;
    int col = idx % 4608, row = idx / 4608;
    int b = row >> 4, pos = row & 15;
    int y = pos >> 2, xx = pos & 3;
    int c = col / 9, rem = col % 9;
    int kh = rem / 3, kw = rem - kh * 3;
    p1[idx] = f2bf(x[((size_t)(b * 512 + c) * 6 + (y + kh)) * 6 + (xx + kw)]);
}

// 3x3 patches of S1g[b][pos(4x4)][c] -> p2p[256 x 4608]
__global__ void pk2p_kernel(const float* __restrict__ S1g, unsigned short* __restrict__ p2p, int total) {
    int idx = blockIdx.x * blockDim.x + threadIdx.x;
    if (idx >= total) return;
    int col = idx % 4608, row = idx / 4608;
    int b = row >> 2, pos = row & 3;
    int y = pos >> 1, xx = pos & 1;
    int c = col / 9, rem = col % 9;
    int kh = rem / 3, kw = rem - kh * 3;
    p2p[idx] = f2bf(S1g[(size_t)(b * 16 + (y + kh) * 4 + (xx + kw)) * 512 + c]);
}

// 2x2 patches of S1g -> p2[576 x 2048]
__global__ void pk2_kernel(const float* __restrict__ S1g, unsigned short* __restrict__ p2, int total) {
    int idx = blockIdx.x * blockDim.x + threadIdx.x;
    if (idx >= total) return;
    int col = idx & 2047, row = idx >> 11;
    int b = row / 9, pos = row - b * 9;
    int y = pos / 3, xx = pos - y * 3;
    int c = col >> 2, rem = col & 3;
    int kh = rem >> 1, kw = rem & 1;
    p2[idx] = f2bf(S1g[(size_t)(b * 16 + (y + kh) * 4 + (xx + kw)) * 512 + c]);
}

// 1x1 patches of S2g -> p3[256 x 512]  (pure convert)
__global__ void pk3_kernel(const float* __restrict__ S2g, unsigned short* __restrict__ p3, int total) {
    int idx = blockIdx.x * blockDim.x + threadIdx.x;
    if (idx >= total) return;
    p3[idx] = f2bf(S2g[idx]);
}

// ---------------------------------------------------------------------------
extern "C" void kernel_launch(void* const* d_in, const int* in_sizes, int n_in,
                              void* d_out, int out_size, void* d_ws, size_t ws_size,
                              hipStream_t stream) {
    const float* x      = (const float*)d_in[0];
    const float* conv_w = (const float*)d_in[1];
    const float* conv_b = (const float*)d_in[2];
    const float* g1_w   = (const float*)d_in[3];
    const float* g1_b   = (const float*)d_in[4];
    const float* g2_w   = (const float*)d_in[5];
    const float* g2_b   = (const float*)d_in[6];
    const float* g3_w   = (const float*)d_in[7];
    const float* g3_b   = (const float*)d_in[8];
    float* out = (float*)d_out;

    char* ws = (char*)d_ws;
    unsigned short* p1  = (unsigned short*)(ws);              //  9,437,184 B  [1024 x 4608] bf16
    float*          S1g = (float*)(ws + 9437184);             //  2,097,152 B  [1024 x 512]  f32
    unsigned short* p2p = (unsigned short*)(ws + 11534336);   //  2,359,296 B  [256 x 4608]  bf16
    unsigned short* p2  = (unsigned short*)(ws + 13893632);   //  2,359,296 B  [576 x 2048]  bf16
    float*          S2g = (float*)(ws + 16252928);            //    524,288 B  [256 x 512]   f32
    unsigned short* p3  = (unsigned short*)(ws + 16777216);   //    262,144 B  [256 x 512]   bf16

    // p1 = extractpatches(x, 3)
    { int total = 1024 * 4608;
      pk1_kernel<<<(total + 255) / 256, 256, 0, stream>>>(x, p1, total); }

    // S1 = relu(conv(x)) as GEMM [1024 x 4608] x [4608 x 512]
    gemm_bf16_relu<<<dim3(8, 4), 256, 0, stream>>>(p1, conv_w, conv_b, S1g,
                                                   1024, 512, 4608, 16, 16, 0, 512);

    // p2' (3x3 patches of S1, for conv2) and p2 (2x2 patches, for kk2)
    { int total = 256 * 4608;
      pk2p_kernel<<<(total + 255) / 256, 256, 0, stream>>>(S1g, p2p, total); }
    { int total = 576 * 2048;
      pk2_kernel<<<(total + 255) / 256, 256, 0, stream>>>(S1g, p2, total); }

    // S2 = relu(conv(S1)) as GEMM [256 x 4608] x [4608 x 512]
    gemm_bf16_relu<<<dim3(2, 4), 256, 0, stream>>>(p2p, conv_w, conv_b, S2g,
                                                   256, 512, 4608, 4, 4, 0, 512);

    // p3 = bf16(S2)
    { int total = 256 * 512;
      pk3_kernel<<<(total + 255) / 256, 256, 0, stream>>>(S2g, p3, total); }

    // kk1: [1024 x 4608] x [4608 x 4608] -> out rows 0..15 per batch
    gemm_bf16_relu<<<dim3(8, 36), 256, 0, stream>>>(p1, g1_w, g1_b, out,
                                                    1024, 4608, 4608, 16, 29, 0, 4608);
    // kk2: [576 x 2048] x [2048 x 4608] -> out rows 16..24 per batch
    gemm_bf16_relu<<<dim3(5, 36), 256, 0, stream>>>(p2, g2_w, g2_b, out,
                                                    576, 4608, 2048, 9, 29, 16, 4608);
    // kk3: [256 x 512] x [512 x 4608] -> out rows 25..28 per batch
    gemm_bf16_relu<<<dim3(2, 36), 256, 0, stream>>>(p3, g3_w, g3_b, out,
                                                    256, 4608, 512, 4, 29, 25, 4608);
}
